// GraphEncoder_67989332296218
// MI455X (gfx1250) — matrix operand order repaired and checked by reading, not compiled, not run on hardware
//
#include <hip/hip_runtime.h>

#define DF 128
#define BN_EPS 1e-5f

typedef __attribute__((ext_vector_type(16))) __bf16 v16bf;
typedef __attribute__((ext_vector_type(8)))  __bf16 v8bf;
typedef __attribute__((ext_vector_type(8)))  float  v8f;

// ---------------- utility kernels ----------------

__global__ __launch_bounds__(256) void k_zero(float* p, int count) {
  int i = blockIdx.x * blockDim.x + threadIdx.x;
  if (i < count) p[i] = 0.0f;
}

__global__ __launch_bounds__(256) void k_deg_init(float* deg, int n) {
  int i = blockIdx.x * blockDim.x + threadIdx.x;
  if (i < n) deg[i] = 1.0f;  // self-loop contributes 1
}

__global__ __launch_bounds__(256) void k_deg_edges(const int* __restrict__ col,
                                                   float* deg, int e) {
  int i = blockIdx.x * blockDim.x + threadIdx.x;
  if (i < e) atomicAdd(&deg[col[i]], 1.0f);
}

__global__ __launch_bounds__(256) void k_deg_rsqrt(float* deg, int n) {
  int i = blockIdx.x * blockDim.x + threadIdx.x;
  if (i < n) deg[i] = rsqrtf(deg[i]);  // deg >= 1 always (self loops)
}

// fp32 -> bf16 hi/lo split (for GEMM inputs)
__global__ __launch_bounds__(256) void k_cvt_split(const float* __restrict__ x,
                                                   __bf16* __restrict__ hi,
                                                   __bf16* __restrict__ lo, int count) {
  int i = blockIdx.x * blockDim.x + threadIdx.x;
  if (i >= count) return;
  float v = x[i];
  __bf16 h = (__bf16)v;
  hi[i] = h;
  lo[i] = (__bf16)(v - (float)h);
}

// W[k][n] fp32 -> transposed bf16 hi/lo WT[n][k] so B-fragment loads are contiguous in K
__global__ __launch_bounds__(256) void k_prepW(const float* __restrict__ W,
                                               __bf16* __restrict__ hiT,
                                               __bf16* __restrict__ loT) {
  int i = blockIdx.x * blockDim.x + threadIdx.x;
  if (i >= DF * DF) return;
  int k = i >> 7, n = i & (DF - 1);
  float w = W[i];
  __bf16 h = (__bf16)w;
  hiT[n * DF + k] = h;
  loT[n * DF + k] = (__bf16)(w - (float)h);
}

// ---------------- WMMA GEMM: H[N,128] = X[N,128] @ W[128,128] ----------------
// One wave computes a 16x128 output tile. hi/lo bf16 split for ~fp32 accuracy:
// X@W ~= Xhi@Whi + Xhi@Wlo + Xlo@Whi (3 WMMAs per 16x16x32 step).
// Both W matrices (64 KB) are staged once per block into LDS in fragment-linear
// order [kt][nt][lane][16 bf16] so the hot loop reads are lane-contiguous
// ds_load_b128 pairs (conflict-free) instead of 8x-redundant global traffic.
__global__ __launch_bounds__(256) void k_gemm(const __bf16* __restrict__ Ahi,
                                              const __bf16* __restrict__ Alo,
                                              const __bf16* __restrict__ BhiT,
                                              const __bf16* __restrict__ BloT,
                                              float* __restrict__ H, int nrows) {
  __shared__ __bf16 sBhi[4 * 8 * 32 * 16];  // 32 KB
  __shared__ __bf16 sBlo[4 * 8 * 32 * 16];  // 32 KB

  // ---- stage W fragments into LDS (all threads; 1024 x 32B per matrix) ----
  for (int e = threadIdx.x; e < 1024; e += 256) {
    const int kt = e >> 8;
    const int nt = (e >> 5) & 7;
    const int ln = e & 31;
    const int bn = nt * 16 + (ln & 15);
    const int kB = kt * 32 + (ln >> 4) * 8;
    const v8bf* sh = (const v8bf*)(BhiT + (size_t)bn * DF + kB);
    v8bf* dh = (v8bf*)(sBhi + e * 16);
    dh[0] = sh[0];
    dh[1] = sh[2];
    const v8bf* sl = (const v8bf*)(BloT + (size_t)bn * DF + kB);
    v8bf* dl = (v8bf*)(sBlo + e * 16);
    dl[0] = sl[0];
    dl[1] = sl[2];
  }
  __syncthreads();

  const int lane = threadIdx.x & 31;
  const int tile = blockIdx.x * (blockDim.x >> 5) + (threadIdx.x >> 5);
  if (tile * 16 >= nrows) return;  // wave-uniform; after barrier so no deadlock
  const int m    = lane & 15;
  const int hi16 = lane >> 4;
  const int arow = tile * 16 + m;

  v8f acc[8];
  for (int nt = 0; nt < 8; ++nt) acc[nt] = {};

  for (int kt = 0; kt < 4; ++kt) {
    // A fragment (16x32 bf16): lanes 0-15 -> K {kt*32+0..7, +16..23}; lanes 16-31 -> +8
    const int kA = kt * 32 + hi16 * 8;
    const __bf16* pa = Ahi + (size_t)arow * DF + kA;
    v8bf a0 = *(const v8bf*)pa;
    v8bf a1 = *(const v8bf*)(pa + 16);
    v16bf ahi = __builtin_shufflevector(a0, a1, 0,1,2,3,4,5,6,7,8,9,10,11,12,13,14,15);
    const __bf16* pl = Alo + (size_t)arow * DF + kA;
    v8bf l0 = *(const v8bf*)pl;
    v8bf l1 = *(const v8bf*)(pl + 16);
    v16bf alo = __builtin_shufflevector(l0, l1, 0,1,2,3,4,5,6,7,8,9,10,11,12,13,14,15);

    for (int nt = 0; nt < 8; ++nt) {
      const int fidx = ((kt * 8 + nt) * 32 + lane) * 16;
      v16bf bhi = *(const v16bf*)(sBhi + fidx);
      v16bf blo = *(const v16bf*)(sBlo + fidx);
      acc[nt] = __builtin_amdgcn_wmma_f32_16x16x32_bf16(false, ahi, false, bhi,
                                                        (short)0, acc[nt], false, false);
      acc[nt] = __builtin_amdgcn_wmma_f32_16x16x32_bf16(false, ahi, false, blo,
                                                        (short)0, acc[nt], false, false);
      acc[nt] = __builtin_amdgcn_wmma_f32_16x16x32_bf16(false, alo, false, bhi,
                                                        (short)0, acc[nt], false, false);
    }
  }

  // C/D layout: VGPR j -> M = j (+8 for lanes 16-31), N = lane&15
  const int rbase = tile * 16 + hi16 * 8;
  for (int nt = 0; nt < 8; ++nt) {
    const int c = nt * 16 + m;
    for (int j = 0; j < 8; ++j) {
      H[(size_t)(rbase + j) * DF + c] = acc[nt][j];
    }
  }
}

// ---------------- aggregation: wave per edge, lane per 4 features ----------------
__global__ __launch_bounds__(256) void k_edge_scatter(const float* __restrict__ H,
                                                      const int* __restrict__ rows,
                                                      const int* __restrict__ cols,
                                                      const float* __restrict__ dinv,
                                                      float* __restrict__ out, int e) {
  const int wave = (int)((blockIdx.x * blockDim.x + threadIdx.x) >> 5);
  const int lane = threadIdx.x & 31;
  if (wave >= e) return;
  const int r = rows[wave];
  const int c = cols[wave];
  const float w = dinv[r] * dinv[c];
  const float4 v = ((const float4*)(H + (size_t)r * DF))[lane];
  float* o = out + (size_t)c * DF + lane * 4;
  atomicAdd(o + 0, v.x * w);
  atomicAdd(o + 1, v.y * w);
  atomicAdd(o + 2, v.z * w);
  atomicAdd(o + 3, v.w * w);
}

// self-loop message + bias: out += H * dinv^2 + b
__global__ __launch_bounds__(256) void k_self_bias(const float* __restrict__ H,
                                                   const float* __restrict__ dinv,
                                                   const float* __restrict__ bias,
                                                   float* __restrict__ out, int total) {
  int i = blockIdx.x * blockDim.x + threadIdx.x;
  if (i >= total) return;
  int row = i >> 7, col = i & (DF - 1);
  float w = dinv[row];
  out[i] += H[i] * (w * w) + bias[col];
}

// ---------------- BatchNorm ----------------
// blockDim = 128 (one thread per column); each block reduces a 64-row strip
__global__ __launch_bounds__(128) void k_bn_stats(const float* __restrict__ h,
                                                  float* __restrict__ stats, int n) {
  const int col = threadIdx.x;
  const int r0 = blockIdx.x * 64;
  const int r1 = (r0 + 64 < n) ? r0 + 64 : n;
  float s = 0.0f, sq = 0.0f;
  for (int r = r0; r < r1; ++r) {
    float v = h[(size_t)r * DF + col];
    s += v;
    sq += v * v;
  }
  atomicAdd(&stats[col], s);
  atomicAdd(&stats[DF + col], sq);
}

// BN apply + ReLU fused with bf16 hi/lo conversion for the next GEMM
__global__ __launch_bounds__(256) void k_bn_relu_cvt(const float* __restrict__ h,
                                                     const float* __restrict__ stats,
                                                     const float* __restrict__ gamma,
                                                     const float* __restrict__ beta,
                                                     __bf16* __restrict__ hi,
                                                     __bf16* __restrict__ lo,
                                                     int n, int total) {
  int i = blockIdx.x * blockDim.x + threadIdx.x;
  if (i >= total) return;
  int col = i & (DF - 1);
  float invN = 1.0f / (float)n;
  float mu  = stats[col] * invN;
  float var = stats[DF + col] * invN - mu * mu;
  float y = (h[i] - mu) * rsqrtf(var + BN_EPS) * gamma[col] + beta[col];
  y = fmaxf(y, 0.0f);
  __bf16 hh = (__bf16)y;
  hi[i] = hh;
  lo[i] = (__bf16)(y - (float)hh);
}

// ---------------- launcher ----------------
extern "C" void kernel_launch(void* const* d_in, const int* in_sizes, int n_in,
                              void* d_out, int out_size, void* d_ws, size_t ws_size,
                              hipStream_t stream) {
  (void)n_in; (void)out_size; (void)ws_size;
  const float* x = (const float*)d_in[0];
  const int*   ei = (const int*)d_in[1];
  const float* Wm[3] = {(const float*)d_in[2], (const float*)d_in[4], (const float*)d_in[6]};
  const float* bm[3] = {(const float*)d_in[3], (const float*)d_in[5], (const float*)d_in[7]};
  const float* gm[2] = {(const float*)d_in[8], (const float*)d_in[10]};
  const float* bt[2] = {(const float*)d_in[9], (const float*)d_in[11]};

  const int N_ = in_sizes[0] / DF;
  const int E_ = in_sizes[1] / 2;
  const int* rows = ei;       // edge_index[0] : gather source
  const int* cols = ei + E_;  // edge_index[1] : scatter target

  auto a256 = [](size_t v) { return (v + 255) & ~(size_t)255; };
  char* ws = (char*)d_ws;
  size_t off = 0;
  float*  dinv = (float*)(ws + off);  off += a256((size_t)N_ * 4);
  __bf16* Xhi  = (__bf16*)(ws + off); off += a256((size_t)N_ * DF * 2);
  __bf16* Xlo  = (__bf16*)(ws + off); off += a256((size_t)N_ * DF * 2);
  float*  Hb   = (float*)(ws + off);  off += a256((size_t)N_ * DF * 4);
  float*  AGG  = (float*)(ws + off);  off += a256((size_t)N_ * DF * 4);
  __bf16* WhiT = (__bf16*)(ws + off); off += a256((size_t)DF * DF * 2);
  __bf16* WloT = (__bf16*)(ws + off); off += a256((size_t)DF * DF * 2);
  float*  stats= (float*)(ws + off);  off += a256(2 * DF * 4);

  const int total = N_ * DF;
  const int TB = 256;

  // symmetric-normalization degrees: deg = 1 (self loop) + in-edge count; dinv = rsqrt(deg)
  k_deg_init <<<(N_ + TB - 1) / TB, TB, 0, stream>>>(dinv, N_);
  k_deg_edges<<<(E_ + TB - 1) / TB, TB, 0, stream>>>(cols, dinv, E_);
  k_deg_rsqrt<<<(N_ + TB - 1) / TB, TB, 0, stream>>>(dinv, N_);

  // layer-1 input in bf16 hi/lo
  k_cvt_split<<<(total + TB - 1) / TB, TB, 0, stream>>>(x, Xhi, Xlo, total);

  const int rowTiles   = (N_ + 15) / 16;
  const int gemmBlocks = (rowTiles + 7) / 8;      // 8 waves per 256-thread block
  const int edgeBlocks = (E_ + 7) / 8;            // wave per edge

  for (int layer = 0; layer < 3; ++layer) {
    float* outb = (layer == 2) ? (float*)d_out : AGG;

    k_prepW<<<(DF * DF + TB - 1) / TB, TB, 0, stream>>>(Wm[layer], WhiT, WloT);
    k_gemm <<<gemmBlocks, TB, 0, stream>>>(Xhi, Xlo, WhiT, WloT, Hb, N_);

    k_zero        <<<(total + TB - 1) / TB, TB, 0, stream>>>(outb, total);
    k_edge_scatter<<<edgeBlocks, TB, 0, stream>>>(Hb, rows, cols, dinv, outb, E_);
    k_self_bias   <<<(total + TB - 1) / TB, TB, 0, stream>>>(Hb, dinv, bm[layer], outb, total);

    if (layer < 2) {
      k_zero       <<<1, 256, 0, stream>>>(stats, 2 * DF);
      k_bn_stats   <<<(N_ + 63) / 64, 128, 0, stream>>>(outb, stats, N_);
      k_bn_relu_cvt<<<(total + TB - 1) / TB, TB, 0, stream>>>(outb, stats, gm[layer],
                                                              bt[layer], Xhi, Xlo, N_, total);
    }
  }
}
